// NGCFUUModel_77214922048057
// MI455X (gfx1250) — matrix operand order repaired and checked by reading, not compiled, not run on hardware
//
#include <hip/hip_runtime.h>
#include <hip/hip_bf16.h>

// ---------------------------------------------------------------------------
// NGCF propagation for MI455X (gfx1250, wave32).
// GEMMs: v_wmma_f32_16x16x32_bf16, f32 accumulate. LDS layouts arranged so
// every WMMA fragment is a contiguous 32B-per-lane region -> ds_load_b128.
// Scatter-add via global_atomic_add_f32. Gather phase is HBM-write bound.
// ---------------------------------------------------------------------------

typedef __attribute__((ext_vector_type(16))) __bf16 v16bf;
typedef __attribute__((ext_vector_type(8)))  __bf16 v8bf;
typedef __attribute__((ext_vector_type(8)))  float  v8f;

#define SHUF16(a, b) __builtin_shufflevector(a, b, 0,1,2,3,4,5,6,7,8,9,10,11,12,13,14,15)

__device__ __forceinline__ __bf16 f2bf(float f) {
    // round-to-nearest-even f32 -> bf16 (bit trick)
    unsigned u = __builtin_bit_cast(unsigned, f);
    u += 0x7FFFu + ((u >> 16) & 1u);
    unsigned short h = (unsigned short)(u >> 16);
    return __builtin_bit_cast(__bf16, h);
}

__device__ __forceinline__ unsigned pack2bf_sw(float lo, float hi) {
    unsigned ul = (unsigned)__builtin_bit_cast(unsigned short, f2bf(lo));
    unsigned uh = (unsigned)__builtin_bit_cast(unsigned short, f2bf(hi));
    return (uh << 16) | ul;
}

__device__ __forceinline__ unsigned pack2bf(float lo, float hi) {
#if __has_builtin(__builtin_amdgcn_cvt_pk_bf16_f32)
    auto p = __builtin_amdgcn_cvt_pk_bf16_f32(lo, hi);
    if constexpr (sizeof(p) == 4) {
        return __builtin_bit_cast(unsigned, p);     // v_cvt_pk_bf16_f32
    } else {
        return pack2bf_sw(lo, hi);
    }
#else
    return pack2bf_sw(lo, hi);
#endif
}

// Position of K index k inside a lane-contiguous A-fragment row.
// Fragment (kb, half) occupies positions [kb*32 + half*16, +16) and holds
// K = kb*32 + half*8 + {0..7, 16..23}  (ISA 7.12.2 16-bit A layout).
__device__ __forceinline__ int permA(int k) {
    int kb = k >> 5, r = k & 31;
    return kb * 32 + ((r >> 3) & 1) * 16 + ((r >> 4) & 1) * 8 + (r & 7);
}

#define LDPAD 136   // row pad (bf16 elems): 272B rows -> 2-way-max bank conflicts

// pack 8 consecutive source floats -> one 16B LDS chunk
__device__ __forceinline__ void store_chunk8(__bf16* dst, float4 f0, float4 f1) {
    uint4 q;
    q.x = pack2bf(f0.x, f0.y);
    q.y = pack2bf(f0.z, f0.w);
    q.z = pack2bf(f1.x, f1.y);
    q.w = pack2bf(f1.z, f1.w);
    *(uint4*)dst = q;
}

// ---------------------------------------------------------------------------
// Kernel 1: seed GU_cat[:, 0:128] = Gu      (GU_cat has row stride 512)
// ---------------------------------------------------------------------------
__global__ void k_init_cat(const float* __restrict__ Gu,
                           float* __restrict__ GU, int U) {
    int idx = blockIdx.x * blockDim.x + threadIdx.x;   // over U*32 float4s
    int total = U * 32;
    if (idx < total) {
        int row = idx >> 5, c = idx & 31;
        float4 v = ((const float4*)Gu)[idx];
        ((float4*)(GU + (size_t)row * 512))[c] = v;
    }
}

// Stage W (f32 [k][n] in global) into LDS transposed+bf16: sW[n][permA(k)].
// Thread handles 8 consecutive k for one n -> one ds_store_b128.
__device__ __forceinline__ void stageW(const float* __restrict__ W,
                                       __bf16 (*sW)[LDPAD], int tid) {
    for (int i = tid; i < 16 * 128; i += 256) {
        int n = i & 127, k0 = (i >> 7) * 8;
        uint4 q;
        q.x = pack2bf(W[(size_t)(k0 + 0) * 128 + n], W[(size_t)(k0 + 1) * 128 + n]);
        q.y = pack2bf(W[(size_t)(k0 + 2) * 128 + n], W[(size_t)(k0 + 3) * 128 + n]);
        q.z = pack2bf(W[(size_t)(k0 + 4) * 128 + n], W[(size_t)(k0 + 5) * 128 + n]);
        q.w = pack2bf(W[(size_t)(k0 + 6) * 128 + n], W[(size_t)(k0 + 7) * 128 + n]);
        *(uint4*)&sW[n][permA(k0)] = q;
    }
}

// ---------------------------------------------------------------------------
// Kernel 2: H1 = X @ W + b ; AGG = H1  (self term of aggregation)
//   32-row M-tiles per block iteration; 8 waves each own 16 output columns
//   and keep two accumulators (two M-subtiles) so B fragments are reused.
// ---------------------------------------------------------------------------
__global__ void __launch_bounds__(256)
k_h1(const float* __restrict__ X, int ldx,
     const float* __restrict__ W, const float* __restrict__ bias,
     float* __restrict__ H1, float* __restrict__ AGG,
     int numRows, int numTiles) {
    __shared__ __bf16 sW[128][LDPAD];   // W^T as bf16: sW[n][k]
    __shared__ __bf16 sA[32][LDPAD];    // A tile, K-permuted for contiguity
    const int tid  = threadIdx.x;
    const int lane = tid & 31;
    const int wave = tid >> 5;

    stageW(W, sW, tid);
    __syncthreads();

    for (int tile = blockIdx.x; tile < numTiles; tile += gridDim.x) {
        const int  row0 = tile * 32;
        const bool full = (row0 + 32 <= numRows);
        {   // stage A: thread t -> row t>>3, K chunk (t&7)*16 .. +15
            int m  = tid >> 3;
            int k0 = (tid & 7) * 16;
            int row = row0 + m;
            if (full) {
                const float4* xp = (const float4*)(X + (size_t)row * ldx + k0);
                store_chunk8(&sA[m][permA(k0)],     xp[0], xp[1]);
                store_chunk8(&sA[m][permA(k0 + 8)], xp[2], xp[3]);
            } else {
                #pragma unroll
                for (int j = 0; j < 16; ++j) {
                    int k = k0 + j;
                    float v = (row < numRows) ? X[(size_t)row * ldx + k] : 0.0f;
                    sA[m][permA(k)] = f2bf(v);
                }
            }
        }
        __syncthreads();

        const int half = lane >> 4;
        const int mrow = lane & 15;
        const int nrow = wave * 16 + (lane & 15);
        v8f c0 = {}, c1 = {};
        #pragma unroll
        for (int kb = 0; kb < 4; ++kb) {
            const int off = kb * 32 + half * 16;
            v16bf b  = SHUF16(*(const v8bf*)&sW[nrow][off],
                              *(const v8bf*)&sW[nrow][off + 8]);
            v16bf a0 = SHUF16(*(const v8bf*)&sA[mrow][off],
                              *(const v8bf*)&sA[mrow][off + 8]);
            v16bf a1 = SHUF16(*(const v8bf*)&sA[16 + mrow][off],
                              *(const v8bf*)&sA[16 + mrow][off + 8]);
            c0 = __builtin_amdgcn_wmma_f32_16x16x32_bf16(
                     false, a0, false, b, (short)0, c0, false, false);
            c1 = __builtin_amdgcn_wmma_f32_16x16x32_bf16(
                     false, a1, false, b, (short)0, c1, false, false);
        }
        const int   n  = wave * 16 + (lane & 15);
        const float bn = bias[n];
        if (full) {
            // constant-offset stores: one base, r*128 folds into IOFFSET
            float* ph = H1  + (size_t)(row0 + 8 * half) * 128 + n;
            float* pa = AGG + (size_t)(row0 + 8 * half) * 128 + n;
            #pragma unroll
            for (int r = 0; r < 8; ++r) {
                float v = c0[r] + bn;
                ph[r * 128] = v;  pa[r * 128] = v;
            }
            #pragma unroll
            for (int r = 0; r < 8; ++r) {
                float v = c1[r] + bn;
                ph[(16 + r) * 128] = v;  pa[(16 + r) * 128] = v;
            }
        } else {
            #pragma unroll
            for (int mt = 0; mt < 2; ++mt) {
                #pragma unroll
                for (int r = 0; r < 8; ++r) {
                    int row = row0 + mt * 16 + r + 8 * half;
                    if (row < numRows) {
                        float v = (mt ? c1[r] : c0[r]) + bn;
                        H1 [(size_t)row * 128 + n] = v;
                        AGG[(size_t)row * 128 + n] = v;
                    }
                }
            }
        }
        __syncthreads();
    }
}

// ---------------------------------------------------------------------------
// Kernel 3: per 32-edge tile: msg = H1[src] + (x[src]*x[dst]) @ W2 + b2,
//           atomically accumulated into AGG[dst].
// ---------------------------------------------------------------------------
__global__ void __launch_bounds__(256)
k_edge(const float* __restrict__ X, int ldx,
       const float* __restrict__ W, const float* __restrict__ bias,
       const float* __restrict__ H1,
       const int* __restrict__ src, const int* __restrict__ dst,
       float* __restrict__ AGG, int numEdges, int numTiles) {
    __shared__ __bf16 sW[128][LDPAD];
    __shared__ __bf16 sA[32][LDPAD];
    __shared__ int sS[32], sD[32];
    const int tid  = threadIdx.x;
    const int lane = tid & 31;
    const int wave = tid >> 5;

    stageW(W, sW, tid);
    __syncthreads();

    for (int tile = blockIdx.x; tile < numTiles; tile += gridDim.x) {
        const int  e0   = tile * 32;
        const bool full = (e0 + 32 <= numEdges);
        if (tid < 32) {
            int e = e0 + tid;
            sS[tid] = (e < numEdges) ? src[e] : -1;
            sD[tid] = (e < numEdges) ? dst[e] : -1;
        }
        // prefetch next tile's edge indices (global_prefetch_b8)
        int tnext = tile + gridDim.x;
        if (tid == 0 && tnext < numTiles) {
            __builtin_prefetch(&src[tnext * 32], 0, 0);
            __builtin_prefetch(&dst[tnext * 32], 0, 0);
        }
        __syncthreads();

        {   // stage A: a[m][k] = x[src][k] * x[dst][k], bf16, K-permuted
            int m  = tid >> 3;
            int k0 = (tid & 7) * 16;
            int s = sS[m], d = sD[m];
            if (full) {
                const float4* xs = (const float4*)(X + (size_t)s * ldx + k0);
                const float4* xd = (const float4*)(X + (size_t)d * ldx + k0);
                float4 p0, p1;
                float4 s0 = xs[0], s1 = xs[1], d0 = xd[0], d1 = xd[1];
                p0.x = s0.x * d0.x; p0.y = s0.y * d0.y; p0.z = s0.z * d0.z; p0.w = s0.w * d0.w;
                p1.x = s1.x * d1.x; p1.y = s1.y * d1.y; p1.z = s1.z * d1.z; p1.w = s1.w * d1.w;
                store_chunk8(&sA[m][permA(k0)], p0, p1);
                float4 s2 = xs[2], s3 = xs[3], d2 = xd[2], d3 = xd[3];
                p0.x = s2.x * d2.x; p0.y = s2.y * d2.y; p0.z = s2.z * d2.z; p0.w = s2.w * d2.w;
                p1.x = s3.x * d3.x; p1.y = s3.y * d3.y; p1.z = s3.z * d3.z; p1.w = s3.w * d3.w;
                store_chunk8(&sA[m][permA(k0 + 8)], p0, p1);
            } else {
                #pragma unroll
                for (int j = 0; j < 16; ++j) {
                    int k = k0 + j;
                    float v = 0.0f;
                    if (s >= 0)
                        v = X[(size_t)s * ldx + k] * X[(size_t)d * ldx + k];
                    sA[m][permA(k)] = f2bf(v);
                }
            }
        }
        __syncthreads();

        const int half = lane >> 4;
        const int mrow = lane & 15;
        const int nrow = wave * 16 + (lane & 15);
        v8f c0 = {}, c1 = {};
        #pragma unroll
        for (int kb = 0; kb < 4; ++kb) {
            const int off = kb * 32 + half * 16;
            v16bf b  = SHUF16(*(const v8bf*)&sW[nrow][off],
                              *(const v8bf*)&sW[nrow][off + 8]);
            v16bf a0 = SHUF16(*(const v8bf*)&sA[mrow][off],
                              *(const v8bf*)&sA[mrow][off + 8]);
            v16bf a1 = SHUF16(*(const v8bf*)&sA[16 + mrow][off],
                              *(const v8bf*)&sA[16 + mrow][off + 8]);
            c0 = __builtin_amdgcn_wmma_f32_16x16x32_bf16(
                     false, a0, false, b, (short)0, c0, false, false);
            c1 = __builtin_amdgcn_wmma_f32_16x16x32_bf16(
                     false, a1, false, b, (short)0, c1, false, false);
        }
        const int   n  = wave * 16 + (lane & 15);
        const float bn = bias[n];
        // hoist index lookups: two contiguous 8-int runs -> wide ds loads
        int sv[16], dv[16];
        #pragma unroll
        for (int j = 0; j < 8; ++j) {
            sv[j]     = sS[8 * half + j];
            sv[8 + j] = sS[16 + 8 * half + j];
            dv[j]     = sD[8 * half + j];
            dv[8 + j] = sD[16 + 8 * half + j];
        }
        if (full) {
            #pragma unroll
            for (int r = 0; r < 8; ++r) {
                float v = c0[r] + H1[(size_t)sv[r] * 128 + n] + bn;
                atomicAdd(&AGG[(size_t)dv[r] * 128 + n], v);
            }
            #pragma unroll
            for (int r = 0; r < 8; ++r) {
                float v = c1[r] + H1[(size_t)sv[8 + r] * 128 + n] + bn;
                atomicAdd(&AGG[(size_t)dv[8 + r] * 128 + n], v);
            }
        } else {
            #pragma unroll
            for (int mt = 0; mt < 2; ++mt) {
                #pragma unroll
                for (int r = 0; r < 8; ++r) {
                    int s = sv[mt * 8 + r], d = dv[mt * 8 + r];
                    if (s >= 0) {
                        float v = (mt ? c1[r] : c0[r]) + H1[(size_t)s * 128 + n] + bn;
                        atomicAdd(&AGG[(size_t)d * 128 + n], v);
                    }
                }
            }
        }
        __syncthreads();
    }
}

// ---------------------------------------------------------------------------
// Kernel 4: x = l2_normalize(leaky_relu(AGG)) -> next column slice of GU_cat
// ---------------------------------------------------------------------------
__global__ void __launch_bounds__(256)
k_norm(const float* __restrict__ AGG, float* __restrict__ GUcol, int U) {
    int row  = blockIdx.x * 8 + (threadIdx.x >> 5);
    int lane = threadIdx.x & 31;
    if (row >= U) return;
    float4 v = ((const float4*)(AGG + (size_t)row * 128))[lane];
    v.x = v.x > 0.0f ? v.x : 0.01f * v.x;
    v.y = v.y > 0.0f ? v.y : 0.01f * v.y;
    v.z = v.z > 0.0f ? v.z : 0.01f * v.z;
    v.w = v.w > 0.0f ? v.w : 0.01f * v.w;
    float ss = v.x * v.x + v.y * v.y + v.z * v.z + v.w * v.w;
    #pragma unroll
    for (int o = 16; o > 0; o >>= 1) ss += __shfl_xor(ss, o, 32);
    float inv = 1.0f / fmaxf(sqrtf(ss), 1e-12f);
    v.x *= inv; v.y *= inv; v.z *= inv; v.w *= inv;
    ((float4*)(GUcol + (size_t)row * 512))[lane] = v;
}

// ---------------------------------------------------------------------------
// Kernel 5: gather + dot. One wave per batch row, float4 streaming.
// ---------------------------------------------------------------------------
__global__ void __launch_bounds__(256)
k_gather(const float* __restrict__ GU, const float* __restrict__ Gi,
         const int* __restrict__ users, const int* __restrict__ items,
         float* __restrict__ xui, float* __restrict__ ogu,
         float* __restrict__ ogi, int B) {
    int b    = blockIdx.x * 8 + (threadIdx.x >> 5);
    int lane = threadIdx.x & 31;
    if (b >= B) return;
    int u = users[b], it = items[b];
    const float4* gu = (const float4*)(GU + (size_t)u  * 512);
    const float4* gi = (const float4*)(Gi + (size_t)it * 512);
    float4* pu = (float4*)(ogu + (size_t)b * 512);
    float4* pi = (float4*)(ogi + (size_t)b * 512);
    float dot = 0.0f;
    #pragma unroll
    for (int r = 0; r < 4; ++r) {
        int idx = r * 32 + lane;            // 128 float4s per row
        float4 a = gu[idx], c = gi[idx];
        pu[idx] = a;
        pi[idx] = c;
        dot += a.x * c.x + a.y * c.y + a.z * c.z + a.w * c.w;
    }
    #pragma unroll
    for (int o = 16; o > 0; o >>= 1) dot += __shfl_xor(dot, o, 32);
    if (lane == 0) xui[b] = dot;
}

static inline int imin(int a, int b) { return a < b ? a : b; }

extern "C" void kernel_launch(void* const* d_in, const int* in_sizes, int n_in,
                              void* d_out, int out_size, void* d_ws, size_t ws_size,
                              hipStream_t stream) {
    const float* Gu   = (const float*)d_in[0];
    const float* Gi   = (const float*)d_in[1];
    const float* W1   = (const float*)d_in[2];
    const float* W2   = (const float*)d_in[3];
    const float* b1   = (const float*)d_in[4];
    const float* b2   = (const float*)d_in[5];
    const int*  eidx  = (const int*)d_in[6];
    const int*  users = (const int*)d_in[7];
    const int*  items = (const int*)d_in[8];

    const int D   = 128;
    const int L   = 3;
    const int U   = in_sizes[0] / D;       // 50000
    const int E   = in_sizes[6] / 2;       // 800000
    const int B   = in_sizes[7];           // 800000
    const int CAT = D * (L + 1);           // 512

    // workspace: GU_cat [U,512] | H1 [U,128] | AGG [U,128]
    float* GU  = (float*)d_ws;
    float* H1  = GU + (size_t)U * CAT;
    float* AGG = H1 + (size_t)U * D;

    const int* src = eidx;
    const int* dst = eidx + E;

    float* xui = (float*)d_out;
    float* ogu = xui + B;
    float* ogi = ogu + (size_t)B * CAT;

    {   // GU_cat[:,0:128] = Gu
        int total = U * 32;
        k_init_cat<<<(total + 255) / 256, 256, 0, stream>>>(Gu, GU, U);
    }

    const int rowTiles  = (U + 31) / 32;
    const int edgeTiles = (E + 31) / 32;

    for (int l = 0; l < L; ++l) {
        const float* Xl = GU + (size_t)l * D;   // column slice, row stride CAT
        k_h1<<<imin(rowTiles, 2048), 256, 0, stream>>>(
            Xl, CAT, W1 + (size_t)l * D * D, b1 + (size_t)l * D,
            H1, AGG, U, rowTiles);
        k_edge<<<imin(edgeTiles, 2048), 256, 0, stream>>>(
            Xl, CAT, W2 + (size_t)l * D * D, b2 + (size_t)l * D,
            H1, src, dst, AGG, E, edgeTiles);
        k_norm<<<(U + 7) / 8, 256, 0, stream>>>(
            AGG, GU + (size_t)(l + 1) * D, U);
    }

    k_gather<<<(B + 7) / 8, 256, 0, stream>>>(
        GU, Gi, users, items, xui, ogu, ogi, B);
}
